// Molormer_446676598855
// MI455X (gfx1250) — matrix operand order repaired and checked by reading, not compiled, not run on hardware
//
#include <hip/hip_runtime.h>
#include <hip/hip_bf16.h>
#include <stdint.h>

// ---------------------------------------------------------------------------
// Molormer encoder for MI455X (gfx1250).
// - All GEMMs in f16 WMMA (f32 accumulate), weights/activations pre-converted
//   to f16 so LDS staging is a raw byte copy.
// - GEMM uses double-buffered LDS tiles; tiles are copied with
//   GLOBAL_LOAD_ASYNC_TO_LDS_B128 (ASYNCcnt) and overlapped with WMMA;
//   a synchronous uint4 path is the fallback if the builtins are absent.
// Workspace requirement: ~420 MB.
// ---------------------------------------------------------------------------

#if __has_builtin(__builtin_amdgcn_global_load_async_to_lds_b128) && \
    __has_builtin(__builtin_amdgcn_s_wait_asynccnt)
#define USE_ASYNC 1
#else
#define USE_ASYNC 0
#endif

typedef __attribute__((ext_vector_type(16))) _Float16 v16h;
typedef __attribute__((ext_vector_type(8)))  _Float16 v8h;
typedef __attribute__((ext_vector_type(8)))  float    v8f;

static constexpr int BATCH = 128;
static constexpr int LSEQ  = 256;   // 255 nodes + graph token
static constexpr int HID   = 512;
static constexpr int INTER = 1024;
static constexpr int NHEAD = 8;
static constexpr int DHEAD = 64;
static constexpr int NSAMP = 30;    // min(5*ceil(ln 256), 256)
static constexpr int MTOK  = BATCH * LSEQ;  // 32768 rows

// ---------------------------------------------------------------------------
// f32 -> f16 conversion (weights, done once per launch)
// ---------------------------------------------------------------------------
__global__ void __launch_bounds__(256) cvt_f16_kernel(
    const float* __restrict__ src, _Float16* __restrict__ dst, int n)
{
  const int i = blockIdx.x * 256 + threadIdx.x;
  if (i < n) dst[i] = (_Float16)src[i];
}

// ---------------------------------------------------------------------------
// Embedding: emits f32 (for residual stream) and f16 (for WMMA GEMM A-operand)
// ---------------------------------------------------------------------------
__global__ void __launch_bounds__(256) embed_kernel(
    const int* __restrict__ node, const int* __restrict__ ind,
    const int* __restrict__ outd,
    const float* __restrict__ node_emb, const float* __restrict__ in_emb,
    const float* __restrict__ out_emb, const float* __restrict__ gtok,
    float* __restrict__ x, _Float16* __restrict__ xh)
{
  const size_t idx = (size_t)blockIdx.x * blockDim.x + threadIdx.x;
  if (idx >= (size_t)BATCH * LSEQ * HID) return;
  const int h = (int)(idx & (HID - 1));
  const int l = (int)((idx >> 9) & (LSEQ - 1));
  const int b = (int)(idx >> 17);
  float val;
  if (l == 0) {
    val = gtok[h];
  } else {
    const int nl = l - 1;
    const int* np = node + ((size_t)b * 255 + nl) * 9;
    float s = 0.0f;
    #pragma unroll
    for (int f = 0; f < 9; ++f) s += node_emb[(size_t)np[f] * HID + h];
    s += in_emb[(size_t)ind[b * 255 + nl] * HID + h];
    s += out_emb[(size_t)outd[b * 255 + nl] * HID + h];
    val = s;
  }
  x[idx] = val;
  xh[idx] = (_Float16)val;
}

// ---------------------------------------------------------------------------
// Tiled WMMA GEMM: C[M,N] = act(A[M,K] * W[N,K]^T + bias[N]), A/W in f16.
// 128x128x32 tiles, 256 threads (8 wave32), each wave -> 64x32 of C.
// Double-buffered LDS; async-to-LDS copies when available.
// ---------------------------------------------------------------------------
static constexpr int GBM = 128, GBN = 128, GBK = 32;
static constexpr int GLD = 40;   // halves per LDS row = 80B (16B aligned)

#if USE_ASYNC
typedef int v4i_ __attribute__((vector_size(16)));
__device__ __forceinline__ void async_b128(const _Float16* g, _Float16* l) {
  __builtin_amdgcn_global_load_async_to_lds_b128(
      (__attribute__((address_space(1))) v4i_*)g,
      (__attribute__((address_space(3))) v4i_*)l, 0, 0);
}
#endif

// Copy one 128x32-half tile: thread -> row (tid>>1), 16-half chunk (tid&1).
__device__ __forceinline__ void stage_tile(const _Float16* __restrict__ gtile,
                                           int ldg, _Float16* __restrict__ s,
                                           int tid)
{
  const int r  = tid >> 1;
  const int c0 = (tid & 1) * 16;
  const _Float16* src = gtile + (size_t)r * ldg + c0;
  _Float16* dst = s + r * GLD + c0;
#if USE_ASYNC
  async_b128(src, dst);
  async_b128(src + 8, dst + 8);
#else
  uint4 d0 = *(const uint4*)src;
  uint4 d1 = *(const uint4*)(src + 8);
  *(uint4*)dst = d0;
  *(uint4*)(dst + 8) = d1;
#endif
}

template <bool RELU, bool OUT_HALF>
__global__ void __launch_bounds__(256) gemm_bias_kernel(
    const _Float16* __restrict__ A,    // [M][K] f16
    const _Float16* __restrict__ W,    // [N][K] f16
    const float* __restrict__ bias,    // [N]
    float* __restrict__ Cf,            // [M][N] f32 out (if !OUT_HALF)
    _Float16* __restrict__ Ch,         // [M][N] f16 out (if OUT_HALF)
    int N, int K)
{
  __shared__ _Float16 sA[2][GBM * GLD];
  __shared__ _Float16 sB[2][GBN * GLD];

  const int tid  = threadIdx.x;
  const int lane = tid & 31;
  const int wave = tid >> 5;     // 0..7
  const int wm   = wave >> 2;    // 0..1 : M offset wm*64
  const int wn   = wave & 3;     // 0..3 : N offset wn*32
  const int bm0  = blockIdx.y * GBM;
  const int bn0  = blockIdx.x * GBN;
  const int ml   = lane & 15;
  const int hi   = lane >> 4;    // 0/1 half-wave
  const int hi8  = hi * 8;

  v8f acc[4][2];
  #pragma unroll
  for (int i = 0; i < 4; ++i)
    #pragma unroll
    for (int j = 0; j < 2; ++j)
      #pragma unroll
      for (int r = 0; r < 8; ++r) acc[i][j][r] = 0.0f;

  const _Float16* Ag = A + (size_t)bm0 * K;
  const _Float16* Wg = W + (size_t)bn0 * K;

  // prologue: stage first k-tile into buffer 0
  stage_tile(Ag, K, sA[0], tid);
  stage_tile(Wg, K, sB[0], tid);

  int buf = 0;
  for (int k0 = 0; k0 < K; k0 += GBK) {
#if USE_ASYNC
    __builtin_amdgcn_s_wait_asynccnt(0);   // tile for `buf` has landed in LDS
#endif
    __syncthreads();

    // kick off next tile into the other buffer while we compute
    if (k0 + GBK < K) {
      stage_tile(Ag + (k0 + GBK), K, sA[buf ^ 1], tid);
      stage_tile(Wg + (k0 + GBK), K, sB[buf ^ 1], tid);
    }

    const _Float16* pAbuf = sA[buf];
    const _Float16* pBbuf = sB[buf];

    // ---- B fragments: lane = N column (fixed), contiguous K run ----
    v16h bf[2];
    #pragma unroll
    for (int j = 0; j < 2; ++j) {
      const _Float16* pb = pBbuf + (wn * 32 + j * 16 + ml) * GLD + hi * 16;
      v8h b0 = *(const v8h*)(pb);
      v8h b1 = *(const v8h*)(pb + 8);
      #pragma unroll
      for (int t = 0; t < 8; ++t) { bf[j][t] = b0[t]; bf[j][8 + t] = b1[t]; }
    }
    // ---- A fragments + 8 WMMAs per wave per k-step ----
    #pragma unroll
    for (int i = 0; i < 4; ++i) {
      const _Float16* pa = pAbuf + (wm * 64 + i * 16 + ml) * GLD;
      v8h a0 = *(const v8h*)(pa + hi * 16);        // K = hi*16 .. hi*16+7
      v8h a1 = *(const v8h*)(pa + hi * 16 + 8);    // K = hi*16+8 .. +7
      v16h af;
      #pragma unroll
      for (int t = 0; t < 8; ++t) { af[t] = a0[t]; af[8 + t] = a1[t]; }
      #pragma unroll
      for (int j = 0; j < 2; ++j) {
        acc[i][j] = __builtin_amdgcn_wmma_f32_16x16x32_f16(
            false, af, false, bf[j], (short)0, acc[i][j], false, false);
      }
    }
    __syncthreads();
    buf ^= 1;
  }

  // ---- epilogue: C/D layout — VGPR r: lanes 0-15 M=r, lanes 16-31 M=8+r ----
  #pragma unroll
  for (int i = 0; i < 4; ++i) {
    #pragma unroll
    for (int j = 0; j < 2; ++j) {
      const int n = bn0 + wn * 32 + j * 16 + ml;
      const float bn_ = bias[n];
      #pragma unroll
      for (int r = 0; r < 8; ++r) {
        const int m = bm0 + wm * 64 + i * 16 + hi8 + r;
        float val = acc[i][j][r] + bn_;
        if (RELU) val = fmaxf(val, 0.0f);
        if (OUT_HALF) Ch[(size_t)m * N + n] = (_Float16)val;
        else          Cf[(size_t)m * N + n] = val;
      }
    }
  }
}

// ---------------------------------------------------------------------------
// ProbSparse metric: M[b,h,l] = max_n(qk) - sum_n(qk)/L over 30 hashed samples
// ---------------------------------------------------------------------------
__device__ __forceinline__ unsigned hash_u32(unsigned x) {
  x ^= x >> 16; x *= 0x7feb352du; x ^= x >> 15; x *= 0x846ca68bu; x ^= x >> 16;
  return x;
}

__global__ void __launch_bounds__(256) metric_kernel(
    const float* __restrict__ q, const float* __restrict__ k,
    float* __restrict__ Mout, unsigned seed)
{
  const int t = blockIdx.x * blockDim.x + threadIdx.x;   // B*NHEAD*LSEQ
  if (t >= BATCH * NHEAD * LSEQ) return;
  const int l = t & (LSEQ - 1);
  const int h = (t >> 8) & (NHEAD - 1);
  const int b = t >> 11;
  const float* qp = q + ((size_t)b * LSEQ + l) * HID + h * DHEAD;
  const size_t kbase = (size_t)b * LSEQ * HID + h * DHEAD;
  float mx = -1e30f, sum = 0.0f;
  for (int n = 0; n < NSAMP; ++n) {
    const int j = (int)(hash_u32(seed + (unsigned)(l * NSAMP + n)) & (LSEQ - 1));
    const float* kp = k + kbase + (size_t)j * HID;
    float d = 0.0f;
    #pragma unroll 8
    for (int i = 0; i < DHEAD; ++i) d += qp[i] * kp[i];
    mx = fmaxf(mx, d);
    sum += d;
  }
  Mout[t] = mx - sum * (1.0f / (float)LSEQ);
}

// ---------------------------------------------------------------------------
// Top-30 per (b,h) by rank counting (ties broken by index => unique ranks)
// ---------------------------------------------------------------------------
__global__ void __launch_bounds__(256) topk_kernel(
    const float* __restrict__ Mm, int* __restrict__ top)
{
  const int bh = blockIdx.x;
  const int t = threadIdx.x;
  __shared__ float sm[LSEQ];
  sm[t] = Mm[(size_t)bh * LSEQ + t];
  __syncthreads();
  const float mv = sm[t];
  int rank = 0;
  for (int j = 0; j < LSEQ; ++j) {
    const float o = sm[j];
    rank += (o > mv) || (o == mv && j < t);
  }
  if (rank < NSAMP) top[bh * NSAMP + rank] = t;
}

// ---------------------------------------------------------------------------
// Attention: ctx = broadcast(mean_l v); for the 30 selected queries compute
// softmax(q.k/8) @ v and scatter into ctx. ctx emitted in f16 [B, L, HID]
// (it is only consumed as the A-operand of the output-projection GEMM).
// ---------------------------------------------------------------------------
__global__ void __launch_bounds__(256) attn_kernel(
    const float* __restrict__ q, const float* __restrict__ k,
    const float* __restrict__ v, const int* __restrict__ top,
    _Float16* __restrict__ ctxh)
{
  const int bh = blockIdx.x;
  const int b = bh >> 3, h = bh & 7;
  const int t = threadIdx.x;
  const size_t base = (size_t)b * LSEQ * HID + (size_t)h * DHEAD;

  __shared__ float vmean[DHEAD];
  __shared__ float sc[LSEQ];
  __shared__ float red[LSEQ];

  if (t < DHEAD) {
    float s = 0.0f;
    for (int l = 0; l < LSEQ; ++l) s += v[base + (size_t)l * HID + t];
    vmean[t] = s * (1.0f / (float)LSEQ);
  }
  __syncthreads();
  for (int e = t; e < LSEQ * DHEAD; e += 256) {
    const int l = e >> 6, d = e & 63;
    ctxh[base + (size_t)l * HID + d] = (_Float16)vmean[d];
  }
  __syncthreads();

  for (int u = 0; u < NSAMP; ++u) {
    const int lq = top[bh * NSAMP + u];
    const float* qp = q + base + (size_t)lq * HID;
    const float* kp = k + base + (size_t)t * HID;
    float s = 0.0f;
    #pragma unroll 8
    for (int i = 0; i < DHEAD; ++i) s += qp[i] * kp[i];
    s *= 0.125f;                      // 1/sqrt(64)
    red[t] = s;
    __syncthreads();
    for (int o = 128; o > 0; o >>= 1) {
      if (t < o) red[t] = fmaxf(red[t], red[t + o]);
      __syncthreads();
    }
    const float mx = red[0];
    __syncthreads();
    const float e = __expf(s - mx);
    red[t] = e;
    __syncthreads();
    for (int o = 128; o > 0; o >>= 1) {
      if (t < o) red[t] += red[t + o];
      __syncthreads();
    }
    const float inv = 1.0f / red[0];
    __syncthreads();
    sc[t] = e * inv;
    __syncthreads();
    if (t < DHEAD) {
      float o = 0.0f;
      for (int l = 0; l < LSEQ; ++l) o += sc[l] * v[base + (size_t)l * HID + t];
      ctxh[base + (size_t)lq * HID + t] = (_Float16)o;
    }
    __syncthreads();
  }
}

// ---------------------------------------------------------------------------
// out = LayerNorm(x (+ y)) * g + beta ; optional parallel f16 copy for GEMMs.
// One 256-thread block per 512-elem row.
// ---------------------------------------------------------------------------
template <bool ADD, bool EMIT_H>
__global__ void __launch_bounds__(256) add_ln_kernel(
    const float* __restrict__ x, const float* __restrict__ y,
    const float* __restrict__ g, const float* __restrict__ beta,
    float* __restrict__ out, _Float16* __restrict__ outh)
{
  const int row = blockIdx.x;
  const int t = threadIdx.x;
  const size_t off = (size_t)row * HID;
  float a0 = x[off + t];
  float a1 = x[off + 256 + t];
  if (ADD) { a0 += y[off + t]; a1 += y[off + 256 + t]; }
  __shared__ float red[256];
  red[t] = a0 + a1;
  __syncthreads();
  for (int o = 128; o > 0; o >>= 1) { if (t < o) red[t] += red[t + o]; __syncthreads(); }
  const float mean = red[0] * (1.0f / (float)HID);
  __syncthreads();
  const float d0 = a0 - mean, d1 = a1 - mean;
  red[t] = d0 * d0 + d1 * d1;
  __syncthreads();
  for (int o = 128; o > 0; o >>= 1) { if (t < o) red[t] += red[t + o]; __syncthreads(); }
  const float rinv = rsqrtf(red[0] * (1.0f / (float)HID) + 1e-5f);
  const float r0 = d0 * rinv * g[t] + beta[t];
  const float r1 = d1 * rinv * g[256 + t] + beta[256 + t];
  out[off + t]       = r0;
  out[off + 256 + t] = r1;
  if (EMIT_H) {
    outh[off + t]       = (_Float16)r0;
    outh[off + 256 + t] = (_Float16)r1;
  }
}

// ---------------------------------------------------------------------------
// Orchestration
// ---------------------------------------------------------------------------
extern "C" void kernel_launch(void* const* d_in, const int* in_sizes, int n_in,
                              void* d_out, int out_size, void* d_ws, size_t ws_size,
                              hipStream_t stream) {
  (void)in_sizes; (void)n_in; (void)out_size; (void)ws_size;

  const int* nodeI[2] = {(const int*)d_in[0], (const int*)d_in[3]};
  const int* indI[2]  = {(const int*)d_in[1], (const int*)d_in[4]};
  const int* outI[2]  = {(const int*)d_in[2], (const int*)d_in[5]};
  const float* node_emb = (const float*)d_in[6];
  const float* in_emb   = (const float*)d_in[7];
  const float* out_emb  = (const float*)d_in[8];
  const float* gtok     = (const float*)d_in[9];
  const float* Wq = (const float*)d_in[10]; const float* bq = (const float*)d_in[11];
  const float* Wk = (const float*)d_in[12]; const float* bk = (const float*)d_in[13];
  const float* Wv = (const float*)d_in[14]; const float* bv = (const float*)d_in[15];
  const float* Wo = (const float*)d_in[16]; const float* bo = (const float*)d_in[17];
  const float* W1 = (const float*)d_in[18]; const float* b1 = (const float*)d_in[19];
  const float* W2 = (const float*)d_in[20]; const float* b2 = (const float*)d_in[21];
  const float* ln1g = (const float*)d_in[22]; const float* ln1b = (const float*)d_in[23];
  const float* ln2g = (const float*)d_in[24]; const float* ln2b = (const float*)d_in[25];
  const float* lnfg = (const float*)d_in[26]; const float* lnfb = (const float*)d_in[27];

  const size_t MH  = (size_t)MTOK * HID;           // 16.78M elements
  const size_t BHL = (size_t)BATCH * NHEAD * LSEQ; // 262144

  float* ws  = (float*)d_ws;
  float* x   = ws;              // f32 residual stream
  float* qb  = x  + MH;
  float* kb  = qb + MH;
  float* vb  = kb + MH;
  float* ao  = vb + MH;         // attn-out / ffn-out (f32)
  float* Mm  = ao + MH;
  int*   top = (int*)(Mm + BHL);

  _Float16* h16  = (_Float16*)(top + 32768);       // 16B-aligned f16 region
  _Float16* x16  = h16;                            // f16 residual stream
  _Float16* c16  = x16 + MH;                       // f16 attention context
  _Float16* wq16 = c16 + MH;
  _Float16* wk16 = wq16 + (size_t)4 * HID * HID;
  _Float16* wv16 = wk16 + (size_t)4 * HID * HID;
  _Float16* wo16 = wv16 + (size_t)4 * HID * HID;
  _Float16* w116 = wo16 + (size_t)4 * HID * HID;
  _Float16* w216 = w116 + (size_t)4 * INTER * HID;
  _Float16* y16  = (_Float16*)qb;  // FFN hidden (f16) reuses dead q+k region

  // ---- convert weights to f16 once per launch ----
  const int nWsm = 4 * HID * HID;     // 1,048,576
  const int nWbg = 4 * INTER * HID;   // 2,097,152
  cvt_f16_kernel<<<nWsm / 256, 256, 0, stream>>>(Wq, wq16, nWsm);
  cvt_f16_kernel<<<nWsm / 256, 256, 0, stream>>>(Wk, wk16, nWsm);
  cvt_f16_kernel<<<nWsm / 256, 256, 0, stream>>>(Wv, wv16, nWsm);
  cvt_f16_kernel<<<nWsm / 256, 256, 0, stream>>>(Wo, wo16, nWsm);
  cvt_f16_kernel<<<nWbg / 256, 256, 0, stream>>>(W1, w116, nWbg);
  cvt_f16_kernel<<<nWbg / 256, 256, 0, stream>>>(W2, w216, nWbg);

  const dim3 gN512(HID / GBN, MTOK / GBM);    // (4, 256)
  const dim3 gN1024(INTER / GBN, MTOK / GBM); // (8, 256)

  for (int g = 0; g < 2; ++g) {
    embed_kernel<<<(BATCH * LSEQ * HID) / 256, 256, 0, stream>>>(
        nodeI[g], indI[g], outI[g], node_emb, in_emb, out_emb, gtok, x, x16);

    for (int l = 0; l < 4; ++l) {
      const size_t wOff  = (size_t)l * HID * HID;
      const size_t w1Off = (size_t)l * INTER * HID;
      gemm_bias_kernel<false, false><<<gN512, 256, 0, stream>>>(
          x16, wq16 + wOff, bq + (size_t)l * HID, qb, nullptr, HID, HID);
      gemm_bias_kernel<false, false><<<gN512, 256, 0, stream>>>(
          x16, wk16 + wOff, bk + (size_t)l * HID, kb, nullptr, HID, HID);
      gemm_bias_kernel<false, false><<<gN512, 256, 0, stream>>>(
          x16, wv16 + wOff, bv + (size_t)l * HID, vb, nullptr, HID, HID);

      const unsigned seed = (unsigned)(g * 4 + l) * 0x9E3779B9u + 0x85ebca6bu;
      metric_kernel<<<(BATCH * NHEAD * LSEQ) / 256, 256, 0, stream>>>(qb, kb, Mm, seed);
      topk_kernel<<<BATCH * NHEAD, 256, 0, stream>>>(Mm, top);
      attn_kernel<<<BATCH * NHEAD, 256, 0, stream>>>(qb, kb, vb, top, c16);

      gemm_bias_kernel<false, false><<<gN512, 256, 0, stream>>>(
          c16, wo16 + wOff, bo + (size_t)l * HID, ao, nullptr, HID, HID);
      add_ln_kernel<true, true><<<MTOK, 256, 0, stream>>>(
          x, ao, ln1g + (size_t)l * HID, ln1b + (size_t)l * HID, x, x16);

      gemm_bias_kernel<true, true><<<gN1024, 256, 0, stream>>>(
          x16, w116 + w1Off, b1 + (size_t)l * INTER, nullptr, y16, INTER, HID);
      gemm_bias_kernel<false, false><<<gN512, 256, 0, stream>>>(
          y16, w216 + w1Off, b2 + (size_t)l * HID, ao, nullptr, HID, INTER);
      add_ln_kernel<true, true><<<MTOK, 256, 0, stream>>>(
          x, ao, ln2g + (size_t)l * HID, ln2b + (size_t)l * HID, x, x16);
    }

    add_ln_kernel<false, false><<<MTOK, 256, 0, stream>>>(
        x, nullptr, lnfg, lnfb, (float*)d_out + (size_t)g * MH, nullptr);
  }
}
